// AttnBlock6_b_16784732193425
// MI455X (gfx1250) — compile-verified, matrix operand order checked by reading, not run on hardware
//
#include <hip/hip_runtime.h>

typedef __attribute__((ext_vector_type(16))) __bf16 v16bf;
typedef __attribute__((ext_vector_type(8)))  float  v8f;
typedef __attribute__((ext_vector_type(4)))  unsigned int v4u;
typedef __attribute__((ext_vector_type(8)))  int v8i;
typedef __attribute__((ext_vector_type(4)))  int v4i;

#define BDIM   8
#define CDIM   256
#define HWSZ   12544      // 112*112
#define PATCHP 196        // patch area
#define PPAD   208        // padded row stride for patch score/prob matrices (16B aligned)
#define DDIM   16384      // C * S
#define AGLOB  3136       // 56*56
#define TGW    56
#define WIMG   112
#define LDK    40         // 32 + 8 pad bf16 elements per LDS row (80B rows, 16B aligned)
#define NSTATBLK 784

__device__ __forceinline__ float  bf2f(__bf16 v){ return (float)v; }
__device__ __forceinline__ __bf16 f2bf(float v){ return (__bf16)v; }

// ---------------- WMMA fragment load from LDS [X][LDK] layout ----------------
// wave32 16-bit A/B layout: lane L -> x = L&15, half g = L>>4;
// elems 0..7  : K = g*8 + 0..7      (VGPR0..3)
// elems 8..15 : K = 16 + g*8 + 0..7 (VGPR4..7)
__device__ __forceinline__ v16bf frag_load(const __bf16* lds, int x, int lane){
  int m = lane & 15, g = lane >> 4;
  const __bf16* p = lds + (size_t)(x + m) * LDK + g * 8;
  union { v4u u[2]; v16bf v; } r;
  r.u[0] = *(const v4u*)(p);        // ds_load_b128
  r.u[1] = *(const v4u*)(p + 16);   // ds_load_b128
  return r.v;
}

// ---------------- TDM staging: global row-major [X][K] tile -> LDS [128][LDK] ----------------
// One TENSOR_LOAD_TO_LDS per tile (issued by wave 0). 2D descriptor:
//   tile_dim0 = 32 elements (64B contiguous per row), tile_dim1 = 128 rows,
//   tensor_dim0/1 = remaining extents (hardware zero-fills OOB reads),
//   LDS padding: pad_interval=3 (every 16 DWORDs = 64B), pad_amount=3 (4 DWORDs = 16B)
//   -> reproduces the 80B-per-row LDK layout natively.
__device__ __forceinline__ void fill_tdm(__bf16* dst, const __bf16* __restrict__ src,
                                         long stride, long r0, long c0,
                                         long Rmax, long Cmax){
  if (threadIdx.x < 32){
    typedef __attribute__((address_space(3))) __bf16 lds_bf16;
    unsigned lds_off = (unsigned)(unsigned long long)(lds_bf16*)dst;
    unsigned long long ga = (unsigned long long)(const char*)(src + r0 * stride + c0);
    unsigned td0 = (unsigned)(Cmax - c0);      // remaining elements along dim0
    unsigned td1 = (unsigned)(Rmax - r0);      // remaining rows
    unsigned long long st = (unsigned long long)stride;
    v4u g0;
    g0[0] = 1u;                                           // count=1, user descriptor
    g0[1] = lds_off;                                      // lds_addr [63:32]
    g0[2] = (unsigned)(ga & 0xFFFFFFFFu);                 // global_addr[31:0]
    g0[3] = (unsigned)((ga >> 32) & 0x01FFFFFFu)          // global_addr[56:32]
          | (2u << 30);                                   // type = 2 ("image")
    v8i g1;
    g1[0] = (int)((1u << 16)                              // data_size = 1 -> 2 bytes
                | (1u << 20)                              // pad_enable
                | (3u << 22)                              // pad_interval: 16 DWORDs
                | (3u << 25));                            // pad_amount: 4 DWORDs
    g1[1] = (int)((td0 & 0xFFFFu) << 16);                 // tensor_dim0[15:0]
    g1[2] = (int)((td0 >> 16) | ((td1 & 0xFFFFu) << 16)); // tensor_dim0[31:16], tensor_dim1[15:0]
    g1[3] = (int)((td1 >> 16) | (32u << 16));             // tensor_dim1[31:16], tile_dim0=32
    g1[4] = (int)(128u);                                  // tile_dim1=128, tile_dim2=0
    g1[5] = (int)(st & 0xFFFFFFFFu);                      // tensor_dim0_stride[31:0]
    g1[6] = (int)((st >> 32) & 0xFFFFu);                  // tensor_dim0_stride[47:32]
    g1[7] = 0;                                            // tensor_dim1_stride = 0
    v4i gz4; gz4[0] = 0; gz4[1] = 0; gz4[2] = 0; gz4[3] = 0;
    v8i gz8;
    #pragma unroll
    for (int e = 0; e < 8; ++e) gz8[e] = 0;
    __builtin_amdgcn_tensor_load_to_lds(g0, g1, gz4, gz4, gz8, 0);
  }
}

// LDS[i][j] = src[(c0+j)*stride + r0+i]   (source row-major [K][X], transpose on store)
template<bool ALIGNED>
__device__ __forceinline__ void fillT(__bf16* dst, const __bf16* __restrict__ src,
                                      long stride, long r0, long c0,
                                      long Xmax, long Kmax, int tid){
  int j0 = tid >> 4;         // 0..15
  int i  = (tid & 15) * 8;   // 0..120
  #pragma unroll
  for (int s = 0; s < 2; ++s){
    int j = j0 + s * 16;
    long k = c0 + j;
    alignas(16) __bf16 tmp[8];
    if (k < Kmax){
      const __bf16* sp = src + k * stride + r0 + i;
      if (ALIGNED && r0 + i + 8 <= Xmax){
        *(v4u*)tmp = *(const v4u*)sp;
      } else {
        #pragma unroll
        for (int e = 0; e < 8; ++e) tmp[e] = (r0 + i + e < Xmax) ? sp[e] : f2bf(0.f);
      }
    } else {
      #pragma unroll
      for (int e = 0; e < 8; ++e) tmp[e] = f2bf(0.f);
    }
    #pragma unroll
    for (int e = 0; e < 8; ++e) dst[(size_t)(i + e) * LDK + j] = tmp[e];
  }
}

// ---------------- tiled GEMM core: 128x128 block, BK=32, 8 waves ----------------
// MODEA/MODEB: 0 = source row-major [X][K] (TDM path); 1 = source row-major [K][X] (fillT)
// ALIGNA/ALIGNB: for fillT sources: stride*2 bytes multiple of 16, base 16B-aligned
template<int MODEA, int MODEB, bool ALIGNA, bool ALIGNB>
__device__ __forceinline__ void gemm_tile(const __bf16* __restrict__ Asrc, long ldA,
                                          const __bf16* __restrict__ Bsrc, long ldB,
                                          long M, long N, long K,
                                          long row0, long col0, v8f acc[2][4]){
  __shared__ __bf16 As[128 * LDK];
  __shared__ __bf16 Bs[128 * LDK];
  int tid  = threadIdx.x;
  int lane = tid & 31;
  int w    = tid >> 5;
  int wm   = w >> 1;    // 0..3 : 32-row strip
  int wn   = w & 1;     // 0..1 : 64-col strip

  for (long k0 = 0; k0 < K; k0 += 32){
    if (MODEA == 0) fill_tdm(As, Asrc, ldA, row0, k0, M, K);
    else            fillT<ALIGNA>(As, Asrc, ldA, row0, k0, M, K, tid);
    if (MODEB == 0) fill_tdm(Bs, Bsrc, ldB, col0, k0, N, K);
    else            fillT<ALIGNB>(Bs, Bsrc, ldB, col0, k0, N, K, tid);

    long kn = k0 + 32;
    if (kn < K){   // speculative prefetch of next K tile (global_prefetch_b8)
      if (MODEA == 1) __builtin_prefetch(Asrc + kn * ldA + row0 + (tid & 127), 0, 1);
      if (MODEB == 1) __builtin_prefetch(Bsrc + kn * ldB + col0 + (tid & 127), 0, 1);
    }
    __builtin_amdgcn_s_wait_tensorcnt(0);   // TDM writes visible (no-op for waves 1..7)
    __syncthreads();

    v16bf a0 = frag_load(As, wm * 32,      lane);
    v16bf a1 = frag_load(As, wm * 32 + 16, lane);
    #pragma unroll
    for (int t = 0; t < 4; ++t){
      v16bf b = frag_load(Bs, wn * 64 + t * 16, lane);
      acc[0][t] = __builtin_amdgcn_wmma_f32_16x16x32_bf16(false, a0, false, b,
                                                          (short)0, acc[0][t], false, false);
      acc[1][t] = __builtin_amdgcn_wmma_f32_16x16x32_bf16(false, a1, false, b,
                                                          (short)0, acc[1][t], false, false);
    }
    __syncthreads();
  }
}

__device__ __forceinline__ void acc_zero(v8f acc[2][4]){
  #pragma unroll
  for (int i = 0; i < 2; ++i)
    #pragma unroll
    for (int t = 0; t < 4; ++t)
      #pragma unroll
      for (int e = 0; e < 8; ++e) acc[i][t][e] = 0.f;
}

// C/D layout: lane L -> n = L&15, rows m = r + 8*(L>>4) for vgpr r=0..7
template<typename F>
__device__ __forceinline__ void epilogue(v8f acc[2][4], long row0, long col0, F f){
  int tid = threadIdx.x, lane = tid & 31, w = tid >> 5, wm = w >> 1, wn = w & 1;
  int n = lane & 15, mh = (lane >> 4) * 8;
  #pragma unroll
  for (int i = 0; i < 2; ++i)
    #pragma unroll
    for (int t = 0; t < 4; ++t){
      long col = col0 + wn * 64 + t * 16 + n;
      #pragma unroll
      for (int r = 0; r < 8; ++r){
        long row = row0 + wm * 32 + i * 16 + mh + r;
        f(row, col, acc[i][t][r]);
      }
    }
}

// ================= stage kernels =================

// GroupNorm stats pass 1: deterministic block partials (no atomics)
__global__ void k_stats(const float* __restrict__ x, float* __restrict__ part){
  int b = blockIdx.y;
  long base = (long)b * CDIM * HWSZ;
  long n = (long)CDIM * HWSZ;                  // 3,211,264 = 784 * 4096
  long s0 = (long)blockIdx.x * 4096;
  float s = 0.f, s2 = 0.f;
  for (long i = s0 + threadIdx.x; i < s0 + 4096 && i < n; i += 256){
    float v = x[base + i]; s += v; s2 += v * v;
  }
  __shared__ float r1[256], r2[256];
  r1[threadIdx.x] = s; r2[threadIdx.x] = s2; __syncthreads();
  for (int st = 128; st > 0; st >>= 1){
    if (threadIdx.x < st){ r1[threadIdx.x] += r1[threadIdx.x + st]; r2[threadIdx.x] += r2[threadIdx.x + st]; }
    __syncthreads();
  }
  if (threadIdx.x == 0){
    part[((long)b * NSTATBLK + blockIdx.x) * 2]     = r1[0];
    part[((long)b * NSTATBLK + blockIdx.x) * 2 + 1] = r2[0];
  }
}

__global__ void k_finalize(const float* __restrict__ part, float* __restrict__ mi){
  int b = blockIdx.x, t = threadIdx.x;
  float s = 0.f, s2 = 0.f;
  for (int i = t; i < NSTATBLK; i += 256){
    s  += part[((long)b * NSTATBLK + i) * 2];
    s2 += part[((long)b * NSTATBLK + i) * 2 + 1];
  }
  __shared__ float r1[256], r2[256];
  r1[t] = s; r2[t] = s2; __syncthreads();
  for (int st = 128; st > 0; st >>= 1){
    if (t < st){ r1[t] += r1[t + st]; r2[t] += r2[t + st]; }
    __syncthreads();
  }
  if (t == 0){
    float cnt = (float)CDIM * (float)HWSZ;
    float mean = r1[0] / cnt;
    float var  = r2[0] / cnt - mean * mean;
    mi[2 * b]     = mean;
    mi[2 * b + 1] = rsqrtf(var + 1e-5f);
  }
}

__global__ void k_wcvt(const float* __restrict__ wq, const float* __restrict__ wk,
                       const float* __restrict__ wv, const float* __restrict__ wp,
                       __bf16* __restrict__ wqkv, __bf16* __restrict__ wproj){
  int i = blockIdx.x * blockDim.x + threadIdx.x;
  const int n1 = 3 * CDIM * CDIM;
  if (i < n1){
    int which = i / (CDIM * CDIM), r = i % (CDIM * CDIM);
    const float* src = (which == 0) ? wq : (which == 1) ? wk : wv;
    wqkv[i] = f2bf(src[r]);
  }
  if (i < CDIM * CDIM) wproj[i] = f2bf(wp[i]);
}

__global__ void k_norm(const float* __restrict__ x, const float* __restrict__ mi,
                       const float* __restrict__ gw, const float* __restrict__ gb,
                       __bf16* __restrict__ xn){
  long total = (long)BDIM * CDIM * HWSZ;
  for (long i = (long)blockIdx.x * blockDim.x + threadIdx.x; i < total;
       i += (long)gridDim.x * blockDim.x){
    long b = i / ((long)CDIM * HWSZ);
    long c = (i / HWSZ) % CDIM;
    xn[i] = f2bf((x[i] - mi[2 * b]) * mi[2 * b + 1] * gw[c] + gb[c]);
  }
}

// fused QKV: [768 x 256] * [256 x 12544] per batch
__global__ void k_qkv(const __bf16* __restrict__ wqkv, const __bf16* __restrict__ xn,
                      const float* __restrict__ bq, const float* __restrict__ bk,
                      const float* __restrict__ bv,
                      __bf16* __restrict__ qb, __bf16* __restrict__ kb, __bf16* __restrict__ vb){
  int b = blockIdx.z;
  long row0 = (long)blockIdx.y * 128;
  long col0 = (long)blockIdx.x * 128;
  const __bf16* B = xn + (long)b * CDIM * HWSZ;
  v8f acc[2][4]; acc_zero(acc);
  gemm_tile<0,1,true,true>(wqkv, CDIM, B, HWSZ, 768, HWSZ, CDIM, row0, col0, acc);
  epilogue(acc, row0, col0, [&](long row, long col, float v){
    if (col < HWSZ){
      int which = (int)(row >> 8), c = (int)(row & 255);
      float bias = (which == 0) ? bq[c] : (which == 1) ? bk[c] : bv[c];
      __bf16* dst = (which == 0) ? qb : (which == 1) ? kb : vb;
      dst[(long)b * CDIM * HWSZ + (long)c * HWSZ + col] = f2bf(v + bias);
    }
  });
}

__global__ void k_pool(const __bf16* __restrict__ src, __bf16* __restrict__ dst){
  long total = (long)BDIM * CDIM * AGLOB;
  for (long i = (long)blockIdx.x * blockDim.x + threadIdx.x; i < total;
       i += (long)gridDim.x * blockDim.x){
    long bc = i / AGLOB, p = i % AGLOB;
    long y = p / TGW, x = p % TGW;
    const __bf16* s = src + bc * HWSZ + (2 * y) * WIMG + 2 * x;
    float v = bf2f(s[0]) + bf2f(s[1]) + bf2f(s[WIMG]) + bf2f(s[WIMG + 1]);
    dst[i] = f2bf(0.25f * v);
  }
}

// patch scores: att[p][q] = (1/128) * sum_d qm[d][p] km[d][q], qm/km: [16384 x 196] row-major
__global__ void k_pattn1(const __bf16* __restrict__ qb, const __bf16* __restrict__ kb,
                         float* __restrict__ attp){
  int b = blockIdx.z;
  long row0 = (long)blockIdx.y * 128, col0 = (long)blockIdx.x * 128;
  const __bf16* A = qb + (long)b * (long)DDIM * PATCHP;
  const __bf16* B = kb + (long)b * (long)DDIM * PATCHP;
  v8f acc[2][4]; acc_zero(acc);
  gemm_tile<1,1,false,false>(A, PATCHP, B, PATCHP, PATCHP, PATCHP, DDIM, row0, col0, acc);
  float* dst = attp + (long)b * PATCHP * PPAD;
  epilogue(acc, row0, col0, [&](long row, long col, float v){
    if (row < PATCHP && col < PATCHP) dst[row * PPAD + col] = v * 0.0078125f;
  });
}

__global__ void k_psoftmax(const float* __restrict__ attp, __bf16* __restrict__ probs){
  int p = blockIdx.x, b = blockIdx.y, t = threadIdx.x;
  const float* row = attp + ((long)b * PATCHP + p) * PPAD;
  __bf16* orow = probs + ((long)b * PATCHP + p) * PPAD;
  float v = (t < PATCHP) ? row[t] : -3.4e38f;
  __shared__ float red[256];
  red[t] = v; __syncthreads();
  for (int st = 128; st > 0; st >>= 1){ if (t < st) red[t] = fmaxf(red[t], red[t + st]); __syncthreads(); }
  float mx = red[0]; __syncthreads();
  float e = (t < PATCHP) ? __expf(v - mx) : 0.f;
  red[t] = e; __syncthreads();
  for (int st = 128; st > 0; st >>= 1){ if (t < st) red[t] += red[t + st]; __syncthreads(); }
  float inv = 1.f / red[0];
  if (t < PATCHP) orow[t] = f2bf(e * inv);
}

// patch apply: h[d][p'] = 0.75 * sum_q vm[d][q] * probs[p'][q]
__global__ void k_pattn2(const __bf16* __restrict__ vb, const __bf16* __restrict__ probs,
                         __bf16* __restrict__ hbf){
  int b = blockIdx.z;
  long row0 = (long)blockIdx.y * 128, col0 = (long)blockIdx.x * 128;
  const __bf16* A = vb + (long)b * (long)DDIM * PATCHP;
  const __bf16* B = probs + (long)b * PATCHP * PPAD;
  v8f acc[2][4]; acc_zero(acc);
  gemm_tile<0,0,false,true>(A, PATCHP, B, PPAD, DDIM, PATCHP, PATCHP, row0, col0, acc);
  __bf16* hb = hbf + (long)b * (long)DDIM * PATCHP;
  epilogue(acc, row0, col0, [&](long row, long col, float v){
    if (col < PATCHP) hb[row * PATCHP + col] = f2bf(0.75f * v);
  });
}

// global scores: att[p][q] = (1/16) * sum_c qp[c][p] kp[c][q], qp/kp: [256 x 3136] row-major
__global__ void k_gattn1(const __bf16* __restrict__ qp, const __bf16* __restrict__ kp,
                         float* __restrict__ attg, int b){
  long row0 = (long)blockIdx.y * 128, col0 = (long)blockIdx.x * 128;
  const __bf16* A = qp + (long)b * CDIM * AGLOB;
  const __bf16* B = kp + (long)b * CDIM * AGLOB;
  v8f acc[2][4]; acc_zero(acc);
  gemm_tile<1,1,true,true>(A, AGLOB, B, AGLOB, AGLOB, AGLOB, CDIM, row0, col0, acc);
  epilogue(acc, row0, col0, [&](long row, long col, float v){
    if (row < AGLOB && col < AGLOB) attg[row * AGLOB + col] = v * 0.0625f;
  });
}

__global__ void k_gsoftmax(const float* __restrict__ attg, __bf16* __restrict__ probs){
  int rowi = blockIdx.x, t = threadIdx.x;
  const float* r = attg + (long)rowi * AGLOB;
  __bf16* o = probs + (long)rowi * AGLOB;
  __shared__ float red[256];
  float mx = -3.4e38f;
  for (int i = t; i < AGLOB; i += 256) mx = fmaxf(mx, r[i]);
  red[t] = mx; __syncthreads();
  for (int st = 128; st > 0; st >>= 1){ if (t < st) red[t] = fmaxf(red[t], red[t + st]); __syncthreads(); }
  mx = red[0]; __syncthreads();
  float s = 0.f;
  for (int i = t; i < AGLOB; i += 256) s += __expf(r[i] - mx);
  red[t] = s; __syncthreads();
  for (int st = 128; st > 0; st >>= 1){ if (t < st) red[t] += red[t + st]; __syncthreads(); }
  float inv = 1.f / red[0];
  for (int i = t; i < AGLOB; i += 256) o[i] = f2bf(__expf(r[i] - mx) * inv);
}

// global apply: hg[c][p'] = sum_q vp[c][q] * probs[p'][q]
__global__ void k_gattn2(const __bf16* __restrict__ vp, const __bf16* __restrict__ probs,
                         float* __restrict__ hg, int b){
  long row0 = (long)blockIdx.y * 128, col0 = (long)blockIdx.x * 128;
  const __bf16* A = vp + (long)b * CDIM * AGLOB;
  v8f acc[2][4]; acc_zero(acc);
  gemm_tile<0,0,true,true>(A, AGLOB, probs, AGLOB, CDIM, AGLOB, AGLOB, row0, col0, acc);
  epilogue(acc, row0, col0, [&](long row, long col, float v){
    if (row < CDIM && col < AGLOB) hg[row * AGLOB + col] = v;
  });
}

// bilinear upsample 56->112 (half-pixel, edge clamp), blend 0.25 into hbf
__global__ void k_upadd(const float* __restrict__ hg, __bf16* __restrict__ hbf, int b){
  long total = (long)CDIM * HWSZ;
  for (long i = (long)blockIdx.x * blockDim.x + threadIdx.x; i < total;
       i += (long)gridDim.x * blockDim.x){
    long c = i / HWSZ, rem = i % HWSZ;
    int y = (int)(rem / WIMG), x = (int)(rem % WIMG);
    float sy = y * 0.5f - 0.25f, sx = x * 0.5f - 0.25f;
    int y0 = (int)floorf(sy), x0 = (int)floorf(sx);
    float wy = sy - (float)y0, wx = sx - (float)x0;
    int y1 = y0 + 1, x1 = x0 + 1;
    y0 = y0 < 0 ? 0 : (y0 > 55 ? 55 : y0);
    y1 = y1 < 0 ? 0 : (y1 > 55 ? 55 : y1);
    x0 = x0 < 0 ? 0 : (x0 > 55 ? 55 : x0);
    x1 = x1 < 0 ? 0 : (x1 > 55 ? 55 : x1);
    const float* g = hg + c * AGLOB;
    float v = (1.f - wy) * ((1.f - wx) * g[y0 * TGW + x0] + wx * g[y0 * TGW + x1])
            +        wy  * ((1.f - wx) * g[y1 * TGW + x0] + wx * g[y1 * TGW + x1]);
    long idx = (long)b * CDIM * HWSZ + i;
    hbf[idx] = f2bf(bf2f(hbf[idx]) + 0.25f * v);
  }
}

// out = x + Wproj * h
__global__ void k_proj(const __bf16* __restrict__ wproj, const __bf16* __restrict__ hbf,
                       const float* __restrict__ x, float* __restrict__ out){
  int b = blockIdx.z;
  long row0 = (long)blockIdx.y * 128, col0 = (long)blockIdx.x * 128;
  const __bf16* B = hbf + (long)b * CDIM * HWSZ;
  v8f acc[2][4]; acc_zero(acc);
  gemm_tile<0,1,true,true>(wproj, CDIM, B, HWSZ, CDIM, HWSZ, CDIM, row0, col0, acc);
  epilogue(acc, row0, col0, [&](long row, long col, float v){
    if (row < CDIM && col < HWSZ){
      long idx = (long)b * CDIM * HWSZ + row * HWSZ + col;
      out[idx] = x[idx] + v;
    }
  });
}

extern "C" void kernel_launch(void* const* d_in, const int* in_sizes, int n_in,
                              void* d_out, int out_size, void* d_ws, size_t ws_size,
                              hipStream_t stream) {
  const float* x  = (const float*)d_in[0];
  const float* gw = (const float*)d_in[1];
  const float* gb = (const float*)d_in[2];
  const float* wq = (const float*)d_in[3];
  const float* bq = (const float*)d_in[4];
  const float* wk = (const float*)d_in[5];
  const float* bk = (const float*)d_in[6];
  const float* wv = (const float*)d_in[7];
  const float* bv = (const float*)d_in[8];
  const float* wp = (const float*)d_in[9];
  float* out = (float*)d_out;

  const long NTOT = (long)BDIM * CDIM * HWSZ;

  char* p = (char*)d_ws;
  auto alloc = [&](size_t bytes) -> void* {
    void* r = (void*)p;
    p += (bytes + 255) & ~(size_t)255;
    return r;
  };
  float*  part   = (float*) alloc((size_t)BDIM * NSTATBLK * 2 * 4);
  float*  mi     = (float*) alloc((size_t)BDIM * 2 * 4);
  __bf16* wqkv   = (__bf16*)alloc((size_t)768 * 256 * 2);
  __bf16* wprojb = (__bf16*)alloc((size_t)256 * 256 * 2);
  __bf16* xn     = (__bf16*)alloc((size_t)NTOT * 2);
  __bf16* qb     = (__bf16*)alloc((size_t)NTOT * 2);
  __bf16* kb     = (__bf16*)alloc((size_t)NTOT * 2);
  __bf16* vb     = (__bf16*)alloc((size_t)NTOT * 2);
  __bf16* qp     = (__bf16*)alloc((size_t)BDIM * CDIM * AGLOB * 2);
  __bf16* kp     = (__bf16*)alloc((size_t)BDIM * CDIM * AGLOB * 2);
  __bf16* vp     = (__bf16*)alloc((size_t)BDIM * CDIM * AGLOB * 2);
  float*  attp   = (float*) alloc((size_t)BDIM * PATCHP * PPAD * 4);
  __bf16* pprobs = (__bf16*)alloc((size_t)BDIM * PATCHP * PPAD * 2);
  __bf16* hbf    = (__bf16*)alloc((size_t)NTOT * 2);
  float*  attg   = (float*) alloc((size_t)AGLOB * AGLOB * 4);   // per-batch reuse
  __bf16* gprobs = (__bf16*)alloc((size_t)AGLOB * AGLOB * 2);   // per-batch reuse
  float*  hg     = (float*) alloc((size_t)CDIM * AGLOB * 4);    // per-batch reuse

  k_stats   <<<dim3(NSTATBLK, BDIM), 256, 0, stream>>>(x, part);
  k_finalize<<<BDIM, 256, 0, stream>>>(part, mi);
  k_wcvt    <<<768, 256, 0, stream>>>(wq, wk, wv, wp, wqkv, wprojb);
  k_norm    <<<8192, 256, 0, stream>>>(x, mi, gw, gb, xn);
  k_qkv     <<<dim3(98, 6, BDIM), 256, 0, stream>>>(wqkv, xn, bq, bk, bv, qb, kb, vb);
  k_pool    <<<4096, 256, 0, stream>>>(qb, qp);
  k_pool    <<<4096, 256, 0, stream>>>(kb, kp);
  k_pool    <<<4096, 256, 0, stream>>>(vb, vp);
  k_pattn1  <<<dim3(2, 2, BDIM), 256, 0, stream>>>(qb, kb, attp);
  k_psoftmax<<<dim3(PATCHP, BDIM), 256, 0, stream>>>(attp, pprobs);
  k_pattn2  <<<dim3(2, 128, BDIM), 256, 0, stream>>>(vb, pprobs, hbf);
  for (int b = 0; b < BDIM; ++b){
    k_gattn1  <<<dim3(25, 25), 256, 0, stream>>>(qp, kp, attg, b);
    k_gsoftmax<<<AGLOB, 256, 0, stream>>>(attg, gprobs);
    k_gattn2  <<<dim3(25, 2), 256, 0, stream>>>(vp, gprobs, hg, b);
    k_upadd   <<<4096, 256, 0, stream>>>(hg, hbf, b);
  }
  k_proj<<<dim3(98, 2, BDIM), 256, 0, stream>>>(wprojb, hbf, x, out);
}
